// SelfAttention_32873679683790
// MI455X (gfx1250) — matrix operand order repaired and checked by reading, not compiled, not run on hardware
//
#include <hip/hip_runtime.h>
#include <cmath>
#include <stdint.h>

// Problem constants (match reference)
#define BB 4
#define LL 4096
#define DD 256
#define AA 64

typedef _Float16 h16;
typedef __attribute__((ext_vector_type(16))) _Float16 v16h;
typedef __attribute__((ext_vector_type(8)))  _Float16 v8h;
typedef __attribute__((ext_vector_type(2)))  _Float16 h2;
typedef __attribute__((ext_vector_type(8)))  float    v8f;

// Workspace layout (units: halfs)
//   xh  : [B*L, D]        converted input (row-major)
//   wt  : [3][A, D]       converted weights, TRANSPOSED (n-major => B-frags contiguous)
//   qh  : [B*L, A]        row-major
//   kh  : [B*L, A]        row-major
//   vt  : [B, A, L]       V transposed (dim-major => PV B-frags contiguous)
#define XH_OFF   0
#define WH_OFF   (BB*LL*DD)
#define QH_OFF   (WH_OFF + 3*DD*AA)
#define KH_OFF   (QH_OFF + BB*LL*AA)
#define VT_OFF   (KH_OFF + BB*LL*AA)

__device__ __forceinline__ v16h cat8(v8h lo, v8h hi) {
  return __builtin_shufflevector(lo, hi, 0, 1, 2, 3, 4, 5, 6, 7,
                                         8, 9, 10, 11, 12, 13, 14, 15);
}

// ---------------------------------------------------------------------------
// Kernel 0: fp32 -> f16 conversion; weights stored transposed [A][D]
// ---------------------------------------------------------------------------
__global__ void attn_cvt_kernel(const float* __restrict__ x,
                                const float* __restrict__ wq,
                                const float* __restrict__ wk,
                                const float* __restrict__ wv,
                                h16* __restrict__ ws) {
  const int NX = BB * LL * DD;
  const int NW = DD * AA;
  const int total = NX + 3 * NW;
  int idx = blockIdx.x * blockDim.x + threadIdx.x;
  int stride = gridDim.x * blockDim.x;
  for (int i = idx; i < total; i += stride) {
    if (i < NX) {
      ws[XH_OFF + i] = (h16)x[i];
    } else {
      int j = i - NX;
      int mat = j / NW;
      int r = j - mat * NW;
      int k = r / AA, n = r - k * AA;       // source [D][A] row-major
      const float* wsrc = (mat == 0) ? wq : (mat == 1) ? wk : wv;
      ws[WH_OFF + mat * NW + n * DD + k] = (h16)wsrc[r];  // dest [A][D]
    }
  }
}

// ---------------------------------------------------------------------------
// Kernel 1: QKV projection.  One block = 16 rows, 4 waves; wave w computes
// output columns [16w,16w+16).  A-frags preloaded from LDS x-tile; B-frags
// are single 32B loads from the transposed weights.  V written transposed.
// ---------------------------------------------------------------------------
__global__ void __launch_bounds__(128) attn_qkv_kernel(const h16* __restrict__ wsc,
                                                       h16* __restrict__ ws) {
  __shared__ h16 xs[16 * DD]; // 8 KB x-tile

  const int tid  = threadIdx.x;
  const int lane = tid & 31;
  const int wid  = tid >> 5;
  const int m    = lane & 15;
  const int hi   = lane >> 4;
  const int col  = wid * 16 + m;
  const int rowbase = blockIdx.x * 16;          // flattened [B*L] row
  const int bat = rowbase / LL;
  const int seq = rowbase - bat * LL;

  { // stage x tile, 16B granularity
    const v8h* src = (const v8h*)(wsc + XH_OFF + (size_t)rowbase * DD);
    v8h* dst = (v8h*)xs;
    for (int e = tid; e < 16 * DD / 8; e += 128) dst[e] = src[e];
  }
  __syncthreads();

  // preload the 8 A-fragments (shared by all three projections)
  v16h afrag[DD / 32];
#pragma unroll
  for (int ks = 0; ks < DD / 32; ++ks) {
    const h16* ap = &xs[m * DD + ks * 32];
    afrag[ks] = cat8(*(const v8h*)(ap + 16 * hi), *(const v8h*)(ap + 16 * hi + 8));
  }

#pragma unroll
  for (int mat = 0; mat < 3; ++mat) {
    const h16* wt = wsc + WH_OFF + (size_t)mat * DD * AA;   // [A][D]
    v8f acc;
#pragma unroll
    for (int i = 0; i < 8; ++i) acc[i] = 0.f;
#pragma unroll
    for (int ks = 0; ks < DD / 32; ++ks) {
      v16h b = *(const v16h*)(wt + (size_t)col * DD + ks * 32 + 16 * hi);
      acc = __builtin_amdgcn_wmma_f32_16x16x32_f16(false, afrag[ks], false, b,
                                                   (short)0, acc, false, false);
    }
    if (mat < 2) {
      h16* outp = ws + (mat == 0 ? QH_OFF : KH_OFF);
#pragma unroll
      for (int v = 0; v < 8; ++v)
        outp[(size_t)(rowbase + v + 8 * hi) * AA + col] = (h16)acc[v];
    } else {
      h16* outp = ws + VT_OFF;                              // [B][A][L]
#pragma unroll
      for (int v = 0; v < 8; ++v)
        outp[((size_t)bat * AA + col) * LL + (seq + v + 8 * hi)] = (h16)acc[v];
    }
  }
}

// ---------------------------------------------------------------------------
// Kernel 2: causal flash attention.  One block = (batch, 64 queries), 4 waves
// x 16 query rows.  K/V tiles staged via async global->LDS b128 copies.
// ---------------------------------------------------------------------------
__global__ void __launch_bounds__(128) attn_fa_kernel(const h16* __restrict__ wsc,
                                                      float* __restrict__ out) {
  __shared__ h16 Ks[32 * AA];     // K tile [key][dim]   4 KB
  __shared__ h16 Vst[AA * 32];    // V tile [dim][key]   4 KB
  __shared__ h16 Ps[4 * 16 * 32]; // per-wave P staging  4 KB

  const int tid  = threadIdx.x;
  const int lane = tid & 31;
  const int wid  = tid >> 5;
  const int m    = lane & 15;
  const int hi   = lane >> 4;

  const int nblk  = LL / 64;
  const int b     = blockIdx.x / nblk;
  const int qbase = (blockIdx.x % nblk) * 64;
  const int qrow0 = qbase + wid * 16;
  const size_t rowoff = (size_t)b * LL;

  const h16* qh = wsc + QH_OFF;
  const h16* kh = wsc + KH_OFF;
  const h16* vt = wsc + VT_OFF;

  // Q A-fragments: two 16x32 chunks over head dim 64 (two 16B loads each)
  v16h qa[2];
#pragma unroll
  for (int c = 0; c < 2; ++c) {
    const h16* qp = qh + (rowoff + qrow0 + m) * AA + c * 32;
    qa[c] = cat8(*(const v8h*)(qp + 16 * hi), *(const v8h*)(qp + 16 * hi + 8));
  }

  float mst[8], lst[8];
  v8f oacc[4];
#pragma unroll
  for (int v = 0; v < 8; ++v) { mst[v] = -1e30f; lst[v] = 0.f; }
#pragma unroll
  for (int nt = 0; nt < 4; ++nt)
#pragma unroll
    for (int v = 0; v < 8; ++v) oacc[nt][v] = 0.f;

  const uint32_t kls = (uint32_t)(uintptr_t)(&Ks[0]);
  const uint32_t vls = (uint32_t)(uintptr_t)(&Vst[0]);

  const int ktmax = qbase / 32 + 1; // inclusive; uniform across block
  for (int kt = 0; kt <= ktmax; ++kt) {
    const int kbase = kt * 32;
    __syncthreads(); // previous tile fully consumed
    // --- async stage K (row-major) and V (pre-transposed) tiles -----------
    {
      const h16* kgp = kh + (rowoff + kbase) * AA;     // 4096 B contiguous
#pragma unroll
      for (int it = 0; it < 2; ++it) {
        int c = tid + it * 128;                        // 16B chunk id 0..255
        uint64_t kga = (uint64_t)(uintptr_t)(kgp + c * 8);
        uint32_t kla = kls + (uint32_t)(c * 16);
        asm volatile("global_load_async_to_lds_b128 %0, %1, off"
                     :: "v"(kla), "v"(kga) : "memory");
        int d  = c >> 2;                               // dim row 0..63
        int cc = c & 3;                                // 8-key sub-chunk
        uint64_t vga = (uint64_t)(uintptr_t)(
            vt + ((size_t)b * AA + d) * LL + kbase + cc * 8);
        uint32_t vla = vls + (uint32_t)(d * 64 + cc * 16);
        asm volatile("global_load_async_to_lds_b128 %0, %1, off"
                     :: "v"(vla), "v"(vga) : "memory");
      }
      asm volatile("s_wait_asynccnt 0x0" ::: "memory");
    }
    __syncthreads();

    // --- S[16q x 32k] = Q * K^T -------------------------------------------
    v8f s0, s1;
#pragma unroll
    for (int v = 0; v < 8; ++v) { s0[v] = 0.f; s1[v] = 0.f; }
#pragma unroll
    for (int c = 0; c < 2; ++c) {
      v16h b0 = *(const v16h*)(&Ks[(0 * 16 + m) * AA + c * 32 + 16 * hi]);
      v16h b1 = *(const v16h*)(&Ks[(1 * 16 + m) * AA + c * 32 + 16 * hi]);
      s0 = __builtin_amdgcn_wmma_f32_16x16x32_f16(false, qa[c], false, b0,
                                                  (short)0, s0, false, false);
      s1 = __builtin_amdgcn_wmma_f32_16x16x32_f16(false, qa[c], false, b1,
                                                  (short)0, s1, false, false);
    }

    // --- causal mask + online softmax -------------------------------------
    float p0e[8], p1e[8];
#pragma unroll
    for (int v = 0; v < 8; ++v) {
      int q   = qrow0 + v + 8 * hi;
      float a0 = (kbase + m      <= q) ? s0[v] : -1e30f;
      float a1 = (kbase + 16 + m <= q) ? s1[v] : -1e30f;
      float t = fmaxf(a0, a1);
      t = fmaxf(t, __shfl_xor(t, 1));
      t = fmaxf(t, __shfl_xor(t, 2));
      t = fmaxf(t, __shfl_xor(t, 4));
      t = fmaxf(t, __shfl_xor(t, 8));
      float mnew  = fmaxf(mst[v], t);
      float scale = __expf(mst[v] - mnew);
      float e0 = __expf(a0 - mnew);
      float e1 = __expf(a1 - mnew);
      float rs = e0 + e1;
      rs += __shfl_xor(rs, 1);
      rs += __shfl_xor(rs, 2);
      rs += __shfl_xor(rs, 4);
      rs += __shfl_xor(rs, 8);
      lst[v] = lst[v] * scale + rs;
      mst[v] = mnew;
#pragma unroll
      for (int nt = 0; nt < 4; ++nt) oacc[nt][v] *= scale;
      p0e[v] = e0; p1e[v] = e1;
    }

    // --- P (C-layout f32) -> LDS f16 -> A-layout fragment ------------------
#pragma unroll
    for (int v = 0; v < 8; ++v) {
      int row = v + 8 * hi;
      Ps[wid * 512 + row * 32 + m]      = (h16)p0e[v];
      Ps[wid * 512 + row * 32 + 16 + m] = (h16)p1e[v];
    }
    asm volatile("s_wait_dscnt 0x0" ::: "memory");
    const h16* pp = &Ps[wid * 512 + m * 32];
    v16h pa = cat8(*(const v8h*)(pp + 16 * hi), *(const v8h*)(pp + 16 * hi + 8));

    // --- O += P * V --------------------------------------------------------
#pragma unroll
    for (int nt = 0; nt < 4; ++nt) {
      v16h bv = *(const v16h*)(&Vst[(nt * 16 + m) * 32 + 16 * hi]);
      oacc[nt] = __builtin_amdgcn_wmma_f32_16x16x32_f16(false, pa, false, bv,
                                                        (short)0, oacc[nt],
                                                        false, false);
    }
  }

  // epilogue: 1/l scaling + fp32 stores
#pragma unroll
  for (int v = 0; v < 8; ++v) {
    int q = qrow0 + v + 8 * hi;
    float inv = 1.0f / lst[v];
    float* op = out + (rowoff + q) * AA;
#pragma unroll
    for (int nt = 0; nt < 4; ++nt) op[nt * 16 + m] = oacc[nt][v] * inv;
  }
}

// ---------------------------------------------------------------------------
// Host launcher
// ---------------------------------------------------------------------------
extern "C" void kernel_launch(void* const* d_in, const int* in_sizes, int n_in,
                              void* d_out, int out_size, void* d_ws, size_t ws_size,
                              hipStream_t stream) {
  const float* x  = (const float*)d_in[0];
  const float* wq = (const float*)d_in[1];
  const float* wk = (const float*)d_in[2];
  const float* wv = (const float*)d_in[3];
  h16* ws = (h16*)d_ws;
  float* out = (float*)d_out;

  const int totcvt = BB * LL * DD + 3 * DD * AA;
  attn_cvt_kernel<<<(totcvt + 255) / 256, 256, 0, stream>>>(x, wq, wk, wv, ws);
  attn_qkv_kernel<<<BB * LL / 16, 128, 0, stream>>>(ws, ws);
  attn_fa_kernel<<<BB * (LL / 64), 128, 0, stream>>>(ws, out);
}
